// MultistepDecoder_58016418234921
// MI455X (gfx1250) — compile-verified, matrix operand order checked by reading, not run on hardware
//
#include <hip/hip_runtime.h>
#include <hip/hip_bf16.h>

typedef __bf16 bf16_t;
typedef __attribute__((ext_vector_type(16))) __bf16 v16bf;
typedef __attribute__((ext_vector_type(4)))  __bf16 v4bf;
typedef __attribute__((ext_vector_type(8)))  float  v8f;

constexpr int P = 128, T = 2048, D = 128, H = 64, G3 = 192, OUTC = 24;

// workspace layout (bytes)
constexpr size_t XPROJ_BYTES = (size_t)P * T * G3 * sizeof(bf16_t); // 100.7 MB
constexpr size_t HS_BYTES    = (size_t)P * T * H  * sizeof(bf16_t); //  33.6 MB
constexpr size_t WIH_ELEMS = (size_t)P * G3 * D;     // 3,145,728
constexpr size_t W1_ELEMS  = (size_t)P * 128 * 64;   // 1,048,576
constexpr size_t W2_ELEMS  = (size_t)P * 64 * 128;   // 1,048,576
constexpr size_t W3_ELEMS  = (size_t)P * 32 * 64;    //   262,144
constexpr size_t W4_ELEMS  = (size_t)P * OUTC * 32;  //    98,304

// ---- weight pre-conversion: f32 -> bf16, one pass, vectorized ---------------
__global__ void __launch_bounds__(256)
cvt_f32_bf16_kernel(const float* __restrict__ src, bf16_t* __restrict__ dst, int n4) {
  int i = blockIdx.x * 256 + threadIdx.x;
  if (i < n4) {
    float4 v = ((const float4*)src)[i];
    v4bf o;
    o[0] = (bf16_t)v.x; o[1] = (bf16_t)v.y; o[2] = (bf16_t)v.z; o[3] = (bf16_t)v.w;
    ((v4bf*)dst)[i] = o;
  }
}

// ---- WMMA fragment loaders -------------------------------------------------
// 16-bit A-matrix 16x32 layout (ISA 7.12.2): lane<16 holds row M=lane with
// K = k0+{0..7} in halves 0..7 and K = k0+{16..23} in halves 8..15;
// lane>=16 holds the same row with K offset by +8.  B is loaded as W (N x K)
// rows in the identical layout (B[k][n] = W[n][k]).

static __device__ inline v16bf frag_from_f32(const float* __restrict__ base, int ld,
                                             int row0, int k0, int lane) {
  int r  = row0 + (lane & 15);
  int kk = k0 + ((lane & 16) ? 8 : 0);
  const float* p = base + (size_t)r * ld + kk;
  v16bf f;
#pragma unroll
  for (int i = 0; i < 8; ++i) f[i] = (bf16_t)p[i];
#pragma unroll
  for (int i = 0; i < 8; ++i) f[i + 8] = (bf16_t)p[16 + i];
  return f;
}

static __device__ inline v16bf frag_from_bf16(const bf16_t* __restrict__ base, int ld,
                                              int row0, int k0, int lane) {
  int r  = row0 + (lane & 15);
  int kk = k0 + ((lane & 16) ? 8 : 0);
  const bf16_t* p = base + (size_t)r * ld + kk;
  v16bf f;
#pragma unroll
  for (int i = 0; i < 8; ++i) f[i] = p[i];
#pragma unroll
  for (int i = 0; i < 8; ++i) f[i + 8] = p[16 + i];
  return f;
}

static __device__ inline v16bf frag_from_bf16_guard(const bf16_t* __restrict__ base, int ld,
                                                    int row0, int k0, int lane, int nrows) {
  int r  = row0 + (lane & 15);
  int kk = k0 + ((lane & 16) ? 8 : 0);
  v16bf f;
  if (r < nrows) {
    const bf16_t* p = base + (size_t)r * ld + kk;
#pragma unroll
    for (int i = 0; i < 8; ++i) f[i] = p[i];
#pragma unroll
    for (int i = 0; i < 8; ++i) f[i + 8] = p[16 + i];
  } else {
#pragma unroll
    for (int i = 0; i < 16; ++i) f[i] = (bf16_t)0.0f;
  }
  return f;
}

// ---- Kernel 1: x_proj[p,t,g] = z[p,t,:] . w_ih[p,g,:] + b_ih[p,g] ----------
// Grid (T/64, P), 128 threads (4 waves); each wave: 16 rows x 192 cols.
// B fragments double-buffered so each load retires under the previous WMMA.
__global__ void __launch_bounds__(128)
xproj_kernel(const float* __restrict__ z, const bf16_t* __restrict__ w_ih_bf,
             const float* __restrict__ b_ih, bf16_t* __restrict__ xproj) {
  const int p    = blockIdx.y;
  const int wave = threadIdx.x >> 5;
  const int lane = threadIdx.x & 31;
  const int t0   = blockIdx.x * 64 + wave * 16;
  const float*  zb = z        + (size_t)p * T * D;
  const bf16_t* wb = w_ih_bf  + (size_t)p * G3 * D;
  v8f zero = {};
  v8f acc[12];
#pragma unroll
  for (int i = 0; i < 12; ++i) acc[i] = zero;

  v16bf a = frag_from_f32(zb, D, t0, 0, lane);
#pragma unroll
  for (int k0 = 0; k0 < D; k0 += 32) {
    // prefetch next K-chunk's A fragment before the WMMA chain
    v16bf anxt = (k0 + 32 < D) ? frag_from_f32(zb, D, t0, k0 + 32, lane) : a;
    v16bf bcur = frag_from_bf16(wb, D, 0, k0, lane);
#pragma unroll
    for (int nt = 0; nt < 12; ++nt) {
      v16bf bnxt = (nt < 11) ? frag_from_bf16(wb, D, (nt + 1) * 16, k0, lane) : bcur;
      acc[nt] = __builtin_amdgcn_wmma_f32_16x16x32_bf16(false, a, false, bcur, (short)0,
                                                        acc[nt], false, false);
      bcur = bnxt;
    }
    a = anxt;
  }
  // C/D layout: vgpr j -> row j (lanes 0-15) / row j+8 (lanes 16-31), col = lane%16
  const float* bias = b_ih + (size_t)p * G3;
  bf16_t* ob = xproj + ((size_t)p * T + t0) * G3;
  const int col  = lane & 15;
  const int rofs = (lane & 16) ? 8 : 0;
#pragma unroll
  for (int nt = 0; nt < 12; ++nt) {
    float bv = bias[nt * 16 + col];
#pragma unroll
    for (int j = 0; j < 8; ++j)
      ob[(size_t)(rofs + j) * G3 + nt * 16 + col] = (bf16_t)(acc[nt][j] + bv);
  }
}

// ---- Kernel 2: sequential GRU scan, one workgroup per part ------------------
// 192 threads: thread g owns gate-row g of w_hh (64 floats in VGPRs).
__global__ void __launch_bounds__(192)
gru_kernel(const bf16_t* __restrict__ xproj, const float* __restrict__ w_hh,
           const float* __restrict__ b_hh, bf16_t* __restrict__ hs) {
  const int p = blockIdx.x;
  const int g = threadIdx.x; // 0..191
  __shared__ __align__(16) float h[H];
  __shared__ float pre[2 * H];
  __shared__ float xn_s[H];
  __shared__ float hn_s[H];

  float4 w[16];
  const float4* wrow = (const float4*)(w_hh + ((size_t)p * G3 + g) * H);
#pragma unroll
  for (int i = 0; i < 16; ++i) w[i] = wrow[i];
  const float bh = b_hh[(size_t)p * G3 + g];
  if (g < H) h[g] = 0.0f;
  __syncthreads();

  const bf16_t* xb = xproj + (size_t)p * T * G3;
  bf16_t*       hb = hs    + (size_t)p * T * H;
  const float4* h4 = (const float4*)h;

  for (int t = 0; t < T; ++t) {
    // pull the pre-activation row a few steps ahead into cache
    __builtin_prefetch(xb + (size_t)((t < T - 8) ? t + 8 : t) * G3 + g, 0, 1);
    float s0 = 0.f, s1 = 0.f, s2 = 0.f, s3 = 0.f;
#pragma unroll
    for (int i = 0; i < 16; ++i) {
      float4 hv = h4[i];
      s0 = fmaf(hv.x, w[i].x, s0);
      s1 = fmaf(hv.y, w[i].y, s1);
      s2 = fmaf(hv.z, w[i].z, s2);
      s3 = fmaf(hv.w, w[i].w, s3);
    }
    const float hp = bh + ((s0 + s1) + (s2 + s3)); // h . w_hh[g] + b_hh[g]
    const float xv = (float)xb[(size_t)t * G3 + g];
    if (g < 2 * H) {
      pre[g] = xv + hp;            // r and z gates: x + h projections
    } else {
      xn_s[g - 2 * H] = xv;        // n gate keeps x and h parts separate
      hn_s[g - 2 * H] = hp;
    }
    __syncthreads();
    if (g < H) {
      const float r    = 1.0f / (1.0f + __expf(-pre[g]));
      const float u    = 1.0f / (1.0f + __expf(-pre[H + g]));
      const float n    = tanhf(xn_s[g] + r * hn_s[g]);
      const float hnew = (1.0f - u) * n + u * h[g];
      h[g] = hnew;
      hb[(size_t)t * H + g] = (bf16_t)hnew;
    }
    __syncthreads();
  }
}

// ---- Kernel 3: fused MLP head 64->128->64->32->24 over hs -------------------
template <int KDIM, int NDIM>
static __device__ inline void layer_bf16(const bf16_t* __restrict__ a_base, int a_ld,
                                         const bf16_t* __restrict__ W,
                                         const float* __restrict__ bias,
                                         bf16_t* __restrict__ outbuf, int lane) {
  constexpr int NT = NDIM / 16;
  v8f zero = {};
  v8f acc[NT];
#pragma unroll
  for (int i = 0; i < NT; ++i) acc[i] = zero;

  v16bf a = frag_from_bf16(a_base, a_ld, 0, 0, lane);
#pragma unroll
  for (int k0 = 0; k0 < KDIM; k0 += 32) {
    v16bf anxt = (k0 + 32 < KDIM) ? frag_from_bf16(a_base, a_ld, 0, k0 + 32, lane) : a;
    v16bf bcur = frag_from_bf16(W, KDIM, 0, k0, lane);
#pragma unroll
    for (int nt = 0; nt < NT; ++nt) {
      v16bf bnxt = (nt < NT - 1) ? frag_from_bf16(W, KDIM, (nt + 1) * 16, k0, lane) : bcur;
      acc[nt] = __builtin_amdgcn_wmma_f32_16x16x32_bf16(false, a, false, bcur, (short)0,
                                                        acc[nt], false, false);
      bcur = bnxt;
    }
    a = anxt;
  }
  const int col  = lane & 15;
  const int rofs = (lane & 16) ? 8 : 0;
#pragma unroll
  for (int nt = 0; nt < NT; ++nt) {
    float bv = bias[nt * 16 + col];
#pragma unroll
    for (int j = 0; j < 8; ++j) {
      float v = acc[nt][j] + bv;
      outbuf[(rofs + j) * NDIM + nt * 16 + col] = (bf16_t)(v > 0.f ? v : 0.f);
    }
  }
}

__global__ void __launch_bounds__(128)
mlp_kernel(const bf16_t* __restrict__ hs,
           const bf16_t* __restrict__ w1, const float* __restrict__ b1,
           const bf16_t* __restrict__ w2, const float* __restrict__ b2,
           const bf16_t* __restrict__ w3, const float* __restrict__ b3,
           const bf16_t* __restrict__ w4, const float* __restrict__ b4,
           float* __restrict__ out) {
  const int p    = blockIdx.y;
  const int wave = threadIdx.x >> 5;
  const int lane = threadIdx.x & 31;
  const int t0   = blockIdx.x * 64 + wave * 16;
  // per-wave private ping-pong activation buffers: no cross-wave barriers needed
  __shared__ bf16_t smem[4][2][16 * 128];
  bf16_t* bufA = smem[wave][0];
  bf16_t* bufB = smem[wave][1];

  const bf16_t* a0 = hs + ((size_t)p * T + t0) * H;
  layer_bf16<64, 128>(a0,   H,   w1 + (size_t)p * 128 * 64,  b1 + (size_t)p * 128, bufA, lane);
  layer_bf16<128, 64>(bufA, 128, w2 + (size_t)p * 64 * 128,  b2 + (size_t)p * 64,  bufB, lane);
  layer_bf16<64, 32> (bufB, 64,  w3 + (size_t)p * 32 * 64,   b3 + (size_t)p * 32,  bufA, lane);

  // final layer 32 -> 24: pad N to 32 with zero-guarded weight rows
  v8f zero = {};
  v8f acc[2] = {zero, zero};
  const bf16_t* W4 = w4 + (size_t)p * OUTC * 32;
  {
    v16bf a = frag_from_bf16(bufA, 32, 0, 0, lane);
    v16bf b0 = frag_from_bf16_guard(W4, 32, 0, 0, lane, OUTC);
    v16bf b1f = frag_from_bf16_guard(W4, 32, 16, 0, lane, OUTC);
    acc[0] = __builtin_amdgcn_wmma_f32_16x16x32_bf16(false, a, false, b0, (short)0,
                                                     acc[0], false, false);
    acc[1] = __builtin_amdgcn_wmma_f32_16x16x32_bf16(false, a, false, b1f, (short)0,
                                                     acc[1], false, false);
  }
  float* ob = out + ((size_t)p * T + t0) * OUTC;
  const float* bias = b4 + (size_t)p * OUTC;
  const int col  = lane & 15;
  const int rofs = (lane & 16) ? 8 : 0;
#pragma unroll
  for (int nt = 0; nt < 2; ++nt) {
    int c = nt * 16 + col;
    if (c < OUTC) {
      float bv = bias[c];
#pragma unroll
      for (int j = 0; j < 8; ++j)
        ob[(size_t)(rofs + j) * OUTC + c] = acc[nt][j] + bv;
    }
  }
}

// ---- Host launcher ----------------------------------------------------------
extern "C" void kernel_launch(void* const* d_in, const int* in_sizes, int n_in,
                              void* d_out, int out_size, void* d_ws, size_t ws_size,
                              hipStream_t stream) {
  const float* z    = (const float*)d_in[0];
  const float* w_ih = (const float*)d_in[1];
  const float* w_hh = (const float*)d_in[2];
  const float* b_ih = (const float*)d_in[3];
  const float* b_hh = (const float*)d_in[4];
  const float* w1   = (const float*)d_in[5];
  const float* b1   = (const float*)d_in[6];
  const float* w2   = (const float*)d_in[7];
  const float* b2   = (const float*)d_in[8];
  const float* w3   = (const float*)d_in[9];
  const float* b3   = (const float*)d_in[10];
  const float* w4   = (const float*)d_in[11];
  const float* b4   = (const float*)d_in[12];
  float* out = (float*)d_out;

  char* ws = (char*)d_ws;
  bf16_t* xproj  = (bf16_t*)ws;                       ws += XPROJ_BYTES;
  bf16_t* hs     = (bf16_t*)ws;                       ws += HS_BYTES;
  bf16_t* wih_bf = (bf16_t*)ws;                       ws += WIH_ELEMS * sizeof(bf16_t);
  bf16_t* w1_bf  = (bf16_t*)ws;                       ws += W1_ELEMS * sizeof(bf16_t);
  bf16_t* w2_bf  = (bf16_t*)ws;                       ws += W2_ELEMS * sizeof(bf16_t);
  bf16_t* w3_bf  = (bf16_t*)ws;                       ws += W3_ELEMS * sizeof(bf16_t);
  bf16_t* w4_bf  = (bf16_t*)ws;

  // one-pass weight conversion to bf16 (tiny vs. the 430 MB main traffic)
  auto cvt = [&](const float* s, bf16_t* d, size_t n) {
    int n4 = (int)(n / 4);
    cvt_f32_bf16_kernel<<<(n4 + 255) / 256, 256, 0, stream>>>(s, d, n4);
  };
  cvt(w_ih, wih_bf, WIH_ELEMS);
  cvt(w1,   w1_bf,  W1_ELEMS);
  cvt(w2,   w2_bf,  W2_ELEMS);
  cvt(w3,   w3_bf,  W3_ELEMS);
  cvt(w4,   w4_bf,  W4_ELEMS);

  dim3 gemm_grid(T / 64, P); // 32 x 128 workgroups, 4 waves each

  xproj_kernel<<<gemm_grid, 128, 0, stream>>>(z, wih_bf, b_ih, xproj);
  gru_kernel<<<P, 192, 0, stream>>>(xproj, w_hh, b_hh, hs);
  mlp_kernel<<<gemm_grid, 128, 0, stream>>>(hs, w1_bf, b1, w2_bf, b2, w3_bf, b3, w4_bf, b4, out);
}
